// Processor_11304353923563
// MI455X (gfx1250) — matrix operand order repaired and checked by reading, not compiled, not run on hardware
//
#include <hip/hip_runtime.h>

// MI455X / gfx1250, wave32.  bf16 WMMA path: v_wmma_f32_16x16x32_bf16
// (bf16 A/B, f32 accumulate; LayerNorm/residual/scatter in exact f32).

typedef float  v8f   __attribute__((ext_vector_type(8)));
typedef __bf16 v4bf  __attribute__((ext_vector_type(4)));
typedef __bf16 v8bf  __attribute__((ext_vector_type(8)));
typedef __bf16 v16bf __attribute__((ext_vector_type(16)));

#define HID   128
#define LNEPS 1e-5f

// Concatenate two 16B bf16 chunks into one 16-element fragment register set.
__device__ __forceinline__ v16bf ld_pair(const __bf16* p0, const __bf16* p1) {
  const v8bf lo = *(const v8bf*)p0;
  const v8bf hi = *(const v8bf*)p1;
  return __builtin_shufflevector(lo, hi, 0, 1, 2, 3, 4, 5, 6, 7, 8, 9, 10, 11,
                                 12, 13, 14, 15);
}

// ---------------------------------------------------------------------------
// NT stacked 16x16 tiles of C = A(NT*16 x K bf16, LDS) * Wt^T + bias.
// Wt is bf16 [128 x K] row-major (pre-transposed weights), so the B-fragment
// (lane=col, chunks [k+8h,+8) and [k+16+8h,+8), matching the A-fragment)
// is two contiguous b128 loads.  A-fragment (16-bit A 16x32 layout): lane=row,
// chunks [k+8h,+8) and [k+16+8h,+8) -> two b128 LDS loads, conflict-free with
// LDA padded so LDA/2 % 64 == 4 words.
// ---------------------------------------------------------------------------
template <int K, int LDA, int NT>
__device__ __forceinline__ void gemm_bf16(const __bf16* __restrict__ As,
                                          const __bf16* __restrict__ Wt,
                                          const float* __restrict__ bias,
                                          v8f* acc, int lane, int colbase) {
  const int half = lane >> 4;
  const int row  = lane & 15;
  const int col  = colbase + row;
  const float bv = bias[col];
  const v8f cinit = {bv, bv, bv, bv, bv, bv, bv, bv};
#pragma unroll
  for (int tdx = 0; tdx < NT; ++tdx) acc[tdx] = cinit;
  const __bf16* bp = Wt + col * K + (half << 3);
  const __bf16* ap = As + row * LDA + (half << 3);
#pragma unroll
  for (int k = 0; k < K; k += 32) {
    const v16bf b = ld_pair(bp + k, bp + k + 16);    // 2x global b128, reused NT times
#pragma unroll
    for (int tdx = 0; tdx < NT; ++tdx) {
      const __bf16* a0 = ap + tdx * 16 * LDA + k;
      const v16bf a = ld_pair(a0, a0 + 16);          // 2x ds b128
      acc[tdx] = __builtin_amdgcn_wmma_f32_16x16x32_bf16(
          false, a, false, b, (short)0, acc[tdx], false, false);
    }
  }
}

// C/D layout: vgpr g, lane L -> row g + 8*(L>>4), col L&15.
__device__ __forceinline__ void store_relu_bf16(__bf16* __restrict__ Hs, int ldh,
                                                const v8f& c, int lane,
                                                int colbase, int rowbase) {
  const int col = colbase + (lane & 15);
  const int r0  = rowbase + ((lane >> 4) << 3);
#pragma unroll
  for (int g = 0; g < 8; ++g)
    Hs[(r0 + g) * ldh + col] = (__bf16)fmaxf(c[g], 0.0f);
}

__device__ __forceinline__ void store_f32(float* __restrict__ Hs, int ldh,
                                          const v8f& c, int lane, int colbase,
                                          int rowbase) {
  const int col = colbase + (lane & 15);
  const int r0  = rowbase + ((lane >> 4) << 3);
#pragma unroll
  for (int g = 0; g < 8; ++g) Hs[(r0 + g) * ldh + col] = c[g];
}

// ---------------------------------------------------------------------------
// Edge update: e = LN(MLP([x[dst]|x[src]|ea])) + ea ; aggr[dst] += e
// 32 edges / block, 8 waves x 16 cols, 2 row-tiles per wave.
// ---------------------------------------------------------------------------
#define E_LDA 392   // 384 + 8 bf16 elements
#define H_LDH 136   // 128 + 8 bf16 elements
#define F_LDH 132   // 128 + 4 f32 elements

__global__ __launch_bounds__(256) void edge_kernel(
    const float* __restrict__ x, const long long* __restrict__ ei, int n_edges,
    float* __restrict__ eattr, float* __restrict__ aggr,
    const __bf16* __restrict__ Wt0, const __bf16* __restrict__ Wt1,
    const __bf16* __restrict__ Wt2, const float* __restrict__ b0,
    const float* __restrict__ b1, const float* __restrict__ b2,
    const float* __restrict__ lbeta, const float* __restrict__ lgamma) {
  __shared__ __align__(16) unsigned char smem[32 * E_LDA * 2 + 2 * 32 * H_LDH * 2];
  __bf16* As = (__bf16*)smem;                    // 32 x 392 bf16 (25088 B)
  float*  H3 = (float*)smem;                     // union: 32 x 132 f32 (16896 B)
  __bf16* H1 = (__bf16*)(smem + 32 * E_LDA * 2); // 32 x 136 bf16
  __bf16* H2 = H1 + 32 * H_LDH;
  __shared__ float sMu[32], sRs[32];
  __shared__ int sSrc[32], sDst[32];

  const int t = threadIdx.x, lane = t & 31, wave = t >> 5;
  const int e0 = blockIdx.x * 32;

  if (t < 32) {
    const int e = min(e0 + t, n_edges - 1);
    sSrc[t] = (int)ei[e];
    sDst[t] = (int)ei[n_edges + e];
  }
  __syncthreads();

  // Stage A = [x[dst] | x[src] | ea] as bf16: 32 rows x 96 float4 reads.
  const float4* x4 = (const float4*)x;
  const float4* e4 = (const float4*)eattr;
  for (int i = t; i < 32 * 96; i += 256) {
    const int r = i / 96, q = i - r * 96;
    const int e = min(e0 + r, n_edges - 1);
    float4 v;
    if (q < 32)      v = x4[sDst[r] * 32 + q];
    else if (q < 64) v = x4[sSrc[r] * 32 + (q - 32)];
    else             v = e4[e * 32 + (q - 64)];
    const v4bf w = {(__bf16)v.x, (__bf16)v.y, (__bf16)v.z, (__bf16)v.w};
    *(v4bf*)(As + r * E_LDA + q * 4) = w;
  }
  __syncthreads();

  const int colbase = wave * 16;
  v8f acc[2];
  gemm_bf16<384, E_LDA, 2>(As, Wt0, b0, acc, lane, colbase);
  store_relu_bf16(H1, H_LDH, acc[0], lane, colbase, 0);
  store_relu_bf16(H1, H_LDH, acc[1], lane, colbase, 16);
  __syncthreads();
  gemm_bf16<128, H_LDH, 2>(H1, Wt1, b1, acc, lane, colbase);
  store_relu_bf16(H2, H_LDH, acc[0], lane, colbase, 0);
  store_relu_bf16(H2, H_LDH, acc[1], lane, colbase, 16);
  __syncthreads();
  gemm_bf16<128, H_LDH, 2>(H2, Wt2, b2, acc, lane, colbase);
  store_f32(H3, F_LDH, acc[0], lane, colbase, 0);   // As dead -> reuse as f32 pre-LN
  store_f32(H3, F_LDH, acc[1], lane, colbase, 16);
  __syncthreads();

  if (t < 32) {                                     // per-row LayerNorm stats (f32)
    float s = 0.f, s2 = 0.f;
    for (int j = 0; j < HID; ++j) {
      const float v = H3[t * F_LDH + j];
      s += v; s2 += v * v;
    }
    const float mu  = s * (1.0f / HID);
    const float var = s2 * (1.0f / HID) - mu * mu;
    sMu[t] = mu;
    sRs[t] = rsqrtf(var + LNEPS);
  }
  __syncthreads();

#pragma unroll
  for (int p = 0; p < 2; ++p) {
    const int r = (t >> 4) + 16 * p, l = t & 15;
    const int e = e0 + r;
    if (e < n_edges) {
      const float mu = sMu[r], rs = sRs[r];
      const int d = sDst[r];
#pragma unroll
      for (int j = 0; j < 8; ++j) {
        const int cx = l + 16 * j;
        const float v = lgamma[cx] * (H3[r * F_LDH + cx] - mu) * rs +
                        lbeta[cx] + eattr[e * HID + cx];   // + residual (f32)
        eattr[e * HID + cx] = v;                           // in-place, own tile
        atomicAdd(&aggr[d * HID + cx], v);                 // global_atomic_add_f32
      }
    }
  }
}

// ---------------------------------------------------------------------------
// Node update: xu = LN(MLP([x | aggr])) + x ; x = has_incoming ? xu : x
// ---------------------------------------------------------------------------
#define N_LDA 264   // 256 + 8 bf16 elements

__global__ __launch_bounds__(256) void node_kernel(
    float* __restrict__ x, const float* __restrict__ aggr,
    const unsigned* __restrict__ flags, int n_nodes,
    const __bf16* __restrict__ Wt0, const __bf16* __restrict__ Wt1,
    const __bf16* __restrict__ Wt2, const float* __restrict__ b0,
    const float* __restrict__ b1, const float* __restrict__ b2,
    const float* __restrict__ lbeta, const float* __restrict__ lgamma) {
  __shared__ __align__(16) unsigned char smem[32 * N_LDA * 2 + 2 * 32 * H_LDH * 2];
  __bf16* As = (__bf16*)smem;                    // 32 x 264 bf16 (16896 B)
  float*  H3 = (float*)smem;                     // union: 32 x 132 f32 (16896 B)
  __bf16* H1 = (__bf16*)(smem + 32 * N_LDA * 2);
  __bf16* H2 = H1 + 32 * H_LDH;
  __shared__ float sMu[32], sRs[32];
  __shared__ unsigned sFlag[32];

  const int t = threadIdx.x, lane = t & 31, wave = t >> 5;
  const int n0 = blockIdx.x * 32;

  if (t < 32) sFlag[t] = flags[min(n0 + t, n_nodes - 1)];

  const float4* x4 = (const float4*)x;
  const float4* a4 = (const float4*)aggr;
  for (int i = t; i < 32 * 64; i += 256) {
    const int r = i >> 6, q = i & 63;
    const int n = min(n0 + r, n_nodes - 1);
    const float4 v = (q < 32) ? x4[n * 32 + q] : a4[n * 32 + (q - 32)];
    const v4bf w = {(__bf16)v.x, (__bf16)v.y, (__bf16)v.z, (__bf16)v.w};
    *(v4bf*)(As + r * N_LDA + q * 4) = w;
  }
  __syncthreads();

  const int colbase = wave * 16;
  v8f acc[2];
  gemm_bf16<256, N_LDA, 2>(As, Wt0, b0, acc, lane, colbase);
  store_relu_bf16(H1, H_LDH, acc[0], lane, colbase, 0);
  store_relu_bf16(H1, H_LDH, acc[1], lane, colbase, 16);
  __syncthreads();
  gemm_bf16<128, H_LDH, 2>(H1, Wt1, b1, acc, lane, colbase);
  store_relu_bf16(H2, H_LDH, acc[0], lane, colbase, 0);
  store_relu_bf16(H2, H_LDH, acc[1], lane, colbase, 16);
  __syncthreads();
  gemm_bf16<128, H_LDH, 2>(H2, Wt2, b2, acc, lane, colbase);
  store_f32(H3, F_LDH, acc[0], lane, colbase, 0);
  store_f32(H3, F_LDH, acc[1], lane, colbase, 16);
  __syncthreads();

  if (t < 32) {
    float s = 0.f, s2 = 0.f;
    for (int j = 0; j < HID; ++j) {
      const float v = H3[t * F_LDH + j];
      s += v; s2 += v * v;
    }
    const float mu  = s * (1.0f / HID);
    const float var = s2 * (1.0f / HID) - mu * mu;
    sMu[t] = mu;
    sRs[t] = rsqrtf(var + LNEPS);
  }
  __syncthreads();

#pragma unroll
  for (int p = 0; p < 2; ++p) {
    const int r = (t >> 4) + 16 * p, l = t & 15;
    const int n = n0 + r;
    if (n < n_nodes && sFlag[r]) {
      const float mu = sMu[r], rs = sRs[r];
#pragma unroll
      for (int j = 0; j < 8; ++j) {
        const int cx = l + 16 * j;
        x[n * HID + cx] = lgamma[cx] * (H3[r * F_LDH + cx] - mu) * rs +
                          lbeta[cx] + x[n * HID + cx];   // + residual (f32)
      }
    }
  }
}

// One-time per launch: W f32 [din x 128] -> Wt bf16 [128 x din] (transposed).
__global__ void wprep_kernel(const float* __restrict__ W,
                             __bf16* __restrict__ Wt, int din) {
  const int i = blockIdx.x * 256 + threadIdx.x;   // over 128*din
  if (i < din * HID) {
    const int c = i / din, k = i - c * din;
    Wt[i] = (__bf16)W[k * HID + c];
  }
}

__global__ void zero_kernel(unsigned* __restrict__ p, long long n) {
  long long i = (long long)blockIdx.x * blockDim.x + threadIdx.x;
  const long long stride = (long long)gridDim.x * blockDim.x;
  for (; i < n; i += stride) p[i] = 0u;
}

__global__ void flag_kernel(const long long* __restrict__ ei, int n_edges,
                            unsigned* __restrict__ flags) {
  const int e = blockIdx.x * blockDim.x + threadIdx.x;
  if (e < n_edges) flags[(int)ei[n_edges + e]] = 1u;
}

// ---------------------------------------------------------------------------
// d_in: [0]=x, [1]=edge_index(int64 2xE), [2]=edge_attr, then params leaves
// per layer: edge{W0,W1,W2,b0,b1,b2,beta,g}, node{...}.
// d_out: [x | edge_attr], updated in place across both passes.
// d_ws:  [aggr N*128 f32 | flags N u32 | bf16 transposed weights]
// ---------------------------------------------------------------------------
extern "C" void kernel_launch(void* const* d_in, const int* in_sizes, int n_in,
                              void* d_out, int out_size, void* d_ws,
                              size_t ws_size, hipStream_t stream) {
  const float*     x_in = (const float*)d_in[0];
  const long long* ei   = (const long long*)d_in[1];
  const float*     e_in = (const float*)d_in[2];
  const int n_nodes = in_sizes[0] / HID;   // 50000
  const int n_edges = in_sizes[2] / HID;   // 600000

  float*    x     = (float*)d_out;
  float*    ea    = (float*)d_out + (size_t)n_nodes * HID;
  float*    aggr  = (float*)d_ws;
  unsigned* flags = (unsigned*)((float*)d_ws + (size_t)n_nodes * HID);
  __bf16*   wbase = (__bf16*)((char*)d_ws + (size_t)n_nodes * HID * 4 +
                              (size_t)n_nodes * 4);

  hipMemcpyAsync(x, x_in, (size_t)n_nodes * HID * sizeof(float),
                 hipMemcpyDeviceToDevice, stream);
  hipMemcpyAsync(ea, e_in, (size_t)n_edges * HID * sizeof(float),
                 hipMemcpyDeviceToDevice, stream);

  zero_kernel<<<2048, 256, 0, stream>>>((unsigned*)d_ws,
                                        (long long)n_nodes * HID + n_nodes);
  flag_kernel<<<(n_edges + 255) / 256, 256, 0, stream>>>(ei, n_edges, flags);

  // bf16+transpose all weights once.  Per-layer bf16 element offsets:
  // edge W0 @0 (384*128), W1 @49152, W2 @65536; node W0 @81920 (256*128),
  // W1 @114688, W2 @131072; layer stride 147456.
  const size_t LSTRIDE = 147456;
  const int   widx[6] = {0, 1, 2, 8, 9, 10};
  const size_t woff[6] = {0, 49152, 65536, 81920, 114688, 131072};
  const int   wdin[6] = {384, 128, 128, 256, 128, 128};
  for (int layer = 0; layer < 2; ++layer)
    for (int j = 0; j < 6; ++j) {
      const float* W = (const float*)d_in[3 + layer * 16 + widx[j]];
      const int n = wdin[j] * HID;
      wprep_kernel<<<(n + 255) / 256, 256, 0, stream>>>(
          W, wbase + layer * LSTRIDE + woff[j], wdin[j]);
    }

  for (int layer = 0; layer < 2; ++layer) {
    const float* p[16];
    for (int i = 0; i < 16; ++i) p[i] = (const float*)d_in[3 + layer * 16 + i];
    const __bf16* wt = wbase + layer * LSTRIDE;
    if (layer)
      zero_kernel<<<2048, 256, 0, stream>>>((unsigned*)aggr,
                                            (long long)n_nodes * HID);
    edge_kernel<<<(n_edges + 31) / 32, 256, 0, stream>>>(
        x, ei, n_edges, ea, aggr, wt + woff[0], wt + woff[1], wt + woff[2],
        p[3], p[4], p[5], p[6], p[7]);
    node_kernel<<<(n_nodes + 31) / 32, 256, 0, stream>>>(
        x, aggr, flags, n_nodes, wt + woff[3], wt + woff[4], wt + woff[5],
        p[11], p[12], p[13], p[14], p[15]);
  }
}
